// PCSSC_90623809946183
// MI455X (gfx1250) — matrix operand order repaired and checked by reading, not compiled, not run on hardware
//
#include <hip/hip_runtime.h>

typedef __attribute__((ext_vector_type(16))) _Float16 v16h;
typedef __attribute__((ext_vector_type(8)))  _Float16 v8h;
typedef __attribute__((ext_vector_type(8)))  float    v8f;

#define B_    32
#define N_    2048
#define KNN_K 16
#define BG    (B_ * N_)          // 65536 groups
#define CENC  384
#define EPSF  1e-5f
#define WAVES 4                  // waves per encoder block
#define GRP   4                  // groups processed concurrently per wave

// ---------------- workspace layout (bytes) ----------------
#define WS_NEIGH   ((size_t)0)
#define NEIGH_BYTES ((size_t)BG * KNN_K * 3 * 4)        // 12,582,912
#define WS_W2H     (WS_NEIGH + NEIGH_BYTES)             // 64*32  f16 = 4096
#define WS_W3H     (WS_W2H + (size_t)64*32*2)           // 128*128 f16 = 32768
#define WS_W4H     (WS_W3H + (size_t)128*128*2)         // 384*128 f16 = 98304
#define WS_W1F     (WS_W4H + (size_t)384*128*2)         // 32*3 f32 (bn1-folded)
#define WS_B1F     (WS_W1F + (size_t)32*3*4)
#define WS_B2F     (WS_B1F + (size_t)32*4)
#define WS_B3F     (WS_B2F + (size_t)64*4)
#define WS_B4F     (WS_B3F + (size_t)128*4)
#define WS_PARAM_BYTES ((size_t)(64*32*2 + 128*128*2 + 384*128*2 + 32*3*4 + 32*4 + 64*4 + 128*4 + 384*4))

// ---------------- LDS layout inside encoder (same order as WS params) ----------------
#define L_W2H   0
#define L_W3H   4096
#define L_W4H   36864
#define L_W1F   135168
#define L_B1F   (L_W1F + 384)
#define L_B2F   (L_B1F + 128)
#define L_B3F   (L_B2F + 256)
#define L_B4F   (L_B3F + 512)                    // ends 137984
#define L_HBUF  138240                           // per (wave,grp) 16x128 f16 = 4096 B
#define L_FBUF  (L_HBUF + WAVES*GRP*4096)
#define LDS_BYTES (L_FBUF + WAVES*GRP*4096)      // 269,312 B  ( < 320 KB per WGP )

// =====================================================================
// Kernel 0: fold BN into weights/biases, convert GEMM weights to f16
// =====================================================================
__global__ __launch_bounds__(256) void fold_convert(
    const float* __restrict__ w1, const float* __restrict__ b1,
    const float* __restrict__ g1, const float* __restrict__ be1,
    const float* __restrict__ m1, const float* __restrict__ v1,
    const float* __restrict__ w2, const float* __restrict__ b2,
    const float* __restrict__ w3, const float* __restrict__ b3,
    const float* __restrict__ g3, const float* __restrict__ be3,
    const float* __restrict__ m3, const float* __restrict__ v3,
    const float* __restrict__ w4, const float* __restrict__ b4,
    char* __restrict__ ws)
{
    int t = blockIdx.x * blockDim.x + threadIdx.x;
    _Float16* w2h = (_Float16*)(ws + WS_W2H);
    _Float16* w3h = (_Float16*)(ws + WS_W3H);
    _Float16* w4h = (_Float16*)(ws + WS_W4H);
    float* w1f = (float*)(ws + WS_W1F);
    float* b1f = (float*)(ws + WS_B1F);
    float* b2f = (float*)(ws + WS_B2F);
    float* b3f = (float*)(ws + WS_B3F);
    float* b4f = (float*)(ws + WS_B4F);

    if (t < 384 * 128) w4h[t] = (_Float16)w4[t];
    if (t < 128 * 128) {
        int o = t >> 7;
        float s = g3[o] * rsqrtf(v3[o] + EPSF);
        w3h[t] = (_Float16)(w3[t] * s);           // bn3 scale folded into w3
    }
    if (t < 64 * 32) w2h[t] = (_Float16)w2[t];
    if (t < 384) b4f[t] = b4[t];
    if (t < 128) {
        float s = g3[t] * rsqrtf(v3[t] + EPSF);
        b3f[t] = (b3[t] - m3[t]) * s + be3[t];    // bn3 shift folded
    }
    if (t < 64) b2f[t] = b2[t];
    if (t < 32) {
        float s = g1[t] * rsqrtf(v1[t] + EPSF);
        b1f[t] = (b1[t] - m1[t]) * s + be1[t];    // bn1 fold
        w1f[t * 3 + 0] = w1[t * 3 + 0] * s;
        w1f[t * 3 + 1] = w1[t * 3 + 1] * s;
        w1f[t * 3 + 2] = w1[t * 3 + 2] * s;
    }
}

// =====================================================================
// Kernel 1: KNN top-16 + gather of center-relative neighbor coords
// =====================================================================
__global__ __launch_bounds__(256) void knn_gather(
    const float* __restrict__ pcd, float* __restrict__ neigh)
{
    __shared__ float px[N_], py[N_], pz[N_];
    int bb    = blockIdx.x >> 3;
    int chunk = blockIdx.x & 7;
    const float* p = pcd + (size_t)bb * N_ * 3;
    for (int i = threadIdx.x; i < N_; i += blockDim.x) {
        px[i] = p[i * 3 + 0]; py[i] = p[i * 3 + 1]; pz[i] = p[i * 3 + 2];
    }
    __syncthreads();

    int q = chunk * 256 + threadIdx.x;
    float qx = px[q], qy = py[q], qz = pz[q];

    float bd[KNN_K]; int bi[KNN_K];
#pragma unroll
    for (int s = 0; s < KNN_K; ++s) { bd[s] = 3.0e38f; bi[s] = 0; }

    for (int j = 0; j < N_; ++j) {
        float dx = px[j] - qx, dy = py[j] - qy, dz = pz[j] - qz;
        float d = dx * dx + dy * dy + dz * dz;
        if (d < bd[KNN_K - 1]) {            // rare insert -> cheap scan loop
            float cd = d; int ci = j;
#pragma unroll
            for (int s = 0; s < KNN_K; ++s) {
                bool sw = cd < bd[s];
                float td = bd[s]; int ti = bi[s];
                if (sw) { bd[s] = cd; bi[s] = ci; cd = td; ci = ti; }
            }
        }
    }
    float* outp = neigh + ((size_t)bb * N_ + q) * (KNN_K * 3);
#pragma unroll
    for (int k = 0; k < KNN_K; ++k) {
        int j = bi[k];
        outp[k * 3 + 0] = px[j] - qx;
        outp[k * 3 + 1] = py[j] - qy;
        outp[k * 3 + 2] = pz[j] - qz;
    }
}

// =====================================================================
// Kernel 2: fused encoder, one wave x GRP groups, WMMA f16->f32
// =====================================================================
__device__ inline v16h cat8(v8h lo, v8h hi) {
    return __builtin_shufflevector(lo, hi, 0,1,2,3,4,5,6,7,8,9,10,11,12,13,14,15);
}
// 16-bit A-matrix 16x32 fragment (contiguous k order, matches B fills):
// lane<16: row=lane, elems 0..15 = K+0..15
// lane>=16: row=lane-16, elems 0..15 = K+16..31
__device__ inline v16h load_a_frag(const _Float16* W, int ldk, int tile, int kbase, int lane) {
    int row  = tile * 16 + (lane & 15);
    int koff = (lane >> 4) * 16;
    const _Float16* p = W + row * ldk + kbase + koff;
    v8h lo = *(const v8h*)p;
    v8h hi = *(const v8h*)(p + 8);
    return cat8(lo, hi);
}

// max-reduce over the 16 lanes of a DPP row (half-wave) via row_ror — pure VALU.
template <int CTRL>
__device__ inline float ror_max(float v) {
    int r = __builtin_amdgcn_update_dpp(__float_as_int(v), __float_as_int(v),
                                        CTRL, 0xf, 0xf, false);
    return fmaxf(v, __int_as_float(r));
}
__device__ inline float rowmax16(float v) {
    v = ror_max<0x128>(v);   // row_ror:8
    v = ror_max<0x124>(v);   // row_ror:4
    v = ror_max<0x122>(v);   // row_ror:2
    v = ror_max<0x121>(v);   // row_ror:1
    return v;
}

// packed f16 pair store (32-bit DS store fed by v_cvt_pk_f16_f32)
__device__ inline void store_h2(_Float16* p, float a, float b) {
    union { _Float16 h[2]; unsigned u; } pk;
    pk.h[0] = (_Float16)a; pk.h[1] = (_Float16)b;
    *(unsigned*)p = pk.u;
}

__global__ __launch_bounds__(WAVES * 32) void encoder_kernel(
    const char* __restrict__ ws, float* __restrict__ out)
{
    extern __shared__ char smem[];
    // Stage all folded params (contiguous in ws) into LDS in one sweep.
    {
        const int* src = (const int*)(ws + WS_W2H);
        int* dst = (int*)smem;
        const int nwords = (int)(WS_PARAM_BYTES / 4);
        for (int i = threadIdx.x; i < nwords; i += blockDim.x) dst[i] = src[i];
    }
    __syncthreads();

    const float*   neigh = (const float*)(ws + WS_NEIGH);
    const _Float16* w2h = (const _Float16*)(smem + L_W2H);
    const _Float16* w3h = (const _Float16*)(smem + L_W3H);
    const _Float16* w4h = (const _Float16*)(smem + L_W4H);
    const float* w1f = (const float*)(smem + L_W1F);
    const float* b1f = (const float*)(smem + L_B1F);
    const float* b2f = (const float*)(smem + L_B2F);
    const float* b3f = (const float*)(smem + L_B3F);
    const float* b4f = (const float*)(smem + L_B4F);

    const int lane = threadIdx.x & 31;
    const int wid  = threadIdx.x >> 5;
    const int col  = lane & 15;      // point index (N dim of WMMA)
    const int hsel = lane >> 4;      // half-wave select

    _Float16* hT[GRP];
    _Float16* fT[GRP];
#pragma unroll
    for (int gg = 0; gg < GRP; ++gg) {
        hT[gg] = (_Float16*)(smem + L_HBUF + (wid * GRP + gg) * 4096);
        fT[gg] = (_Float16*)(smem + L_FBUF + (wid * GRP + gg) * 4096);
    }

    const long stride = (long)gridDim.x * WAVES * GRP;
    for (long g0 = ((long)blockIdx.x * WAVES + wid) * GRP; g0 < BG; g0 += stride) {
        // prefetch next iteration's neighbor data while we compute
        __builtin_prefetch(neigh + (g0 + stride) * (KNN_K * 3), 0, 1);

        // ---- conv1 + bn1 + relu: build GRP 32x16 B-fragments directly ----
        v16h f1[GRP];
#pragma unroll
        for (int gg = 0; gg < GRP; ++gg) {
            const float* nb = neigh + (g0 + gg) * (KNN_K * 3) + col * 3;
            float x0 = nb[0], x1 = nb[1], x2 = nb[2];
#pragma unroll
            for (int e = 0; e < 16; ++e) {
                int c = hsel * 16 + e;
                float v = fmaf(w1f[c*3+0], x0,
                           fmaf(w1f[c*3+1], x1, fmaf(w1f[c*3+2], x2, b1f[c])));
                f1[gg][e] = (_Float16)fmaxf(v, 0.0f);
            }
        }

        // ---- conv2 (64x32) : load each A frag once, 4 WMMAs against it ----
#pragma unroll
        for (int t = 0; t < 4; ++t) {
            v16h a = load_a_frag(w2h, 32, t, 0, lane);
            v8f acc[GRP];
#pragma unroll
            for (int gg = 0; gg < GRP; ++gg) {
                acc[gg] = (v8f){};
                acc[gg] = __builtin_amdgcn_wmma_f32_16x16x32_f16(
                    false, a, false, f1[gg], (short)0, acc[gg], false, false);
            }
#pragma unroll
            for (int gg = 0; gg < GRP; ++gg) {
                float vv[8], mm[8];
#pragma unroll
                for (int r = 0; r < 8; ++r) {
                    int row = t * 16 + hsel * 8 + r;
                    vv[r] = acc[gg][r] + b2f[row];
                    mm[r] = rowmax16(vv[r]);
                }
#pragma unroll
                for (int r = 0; r < 8; r += 2) {
                    int row = t * 16 + hsel * 8 + r;
                    store_h2(&hT[gg][col * 128 + row],      mm[r], mm[r + 1]); // fg
                    store_h2(&hT[gg][col * 128 + 64 + row], vv[r], vv[r + 1]); // f2
                }
            }
        }

        // ---- conv3 B fragments for all groups ----
        v16h hb[GRP][4];
#pragma unroll
        for (int gg = 0; gg < GRP; ++gg)
#pragma unroll
            for (int kk = 0; kk < 4; ++kk)
                hb[gg][kk] = *(const v16h*)&hT[gg][col * 128 + kk * 32 + hsel * 16];

        // ---- conv3 (128x128) + folded bn3 + relu -> fT ----
        for (int t = 0; t < 8; ++t) {
            v8f acc[GRP];
#pragma unroll
            for (int gg = 0; gg < GRP; ++gg) acc[gg] = (v8f){};
#pragma unroll
            for (int kk = 0; kk < 4; ++kk) {
                v16h a = load_a_frag(w3h, 128, t, kk * 32, lane);
#pragma unroll
                for (int gg = 0; gg < GRP; ++gg)
                    acc[gg] = __builtin_amdgcn_wmma_f32_16x16x32_f16(
                        false, a, false, hb[gg][kk], (short)0, acc[gg], false, false);
            }
#pragma unroll
            for (int gg = 0; gg < GRP; ++gg)
#pragma unroll
                for (int r = 0; r < 8; r += 2) {
                    int row = t * 16 + hsel * 8 + r;
                    float a0 = fmaxf(acc[gg][r]     + b3f[row],     0.0f);
                    float a1 = fmaxf(acc[gg][r + 1] + b3f[row + 1], 0.0f);
                    store_h2(&fT[gg][col * 128 + row], a0, a1);
                }
        }

        // ---- conv4 B fragments ----
        v16h fb[GRP][4];
#pragma unroll
        for (int gg = 0; gg < GRP; ++gg)
#pragma unroll
            for (int kk = 0; kk < 4; ++kk)
                fb[gg][kk] = *(const v16h*)&fT[gg][col * 128 + kk * 32 + hsel * 16];

        // ---- conv4 (384x128) + DPP rowwise max -> staged in (reused) hT ----
        for (int t = 0; t < 24; ++t) {
            v8f acc[GRP];
#pragma unroll
            for (int gg = 0; gg < GRP; ++gg) acc[gg] = (v8f){};
#pragma unroll
            for (int kk = 0; kk < 4; ++kk) {
                v16h a = load_a_frag(w4h, 128, t, kk * 32, lane);
#pragma unroll
                for (int gg = 0; gg < GRP; ++gg)
                    acc[gg] = __builtin_amdgcn_wmma_f32_16x16x32_f16(
                        false, a, false, fb[gg][kk], (short)0, acc[gg], false, false);
            }
#pragma unroll
            for (int gg = 0; gg < GRP; ++gg) {
                float mv[8];
#pragma unroll
                for (int r = 0; r < 8; ++r) {
                    int row = t * 16 + hsel * 8 + r;
                    mv[r] = rowmax16(acc[gg][r]) + b4f[row];
                }
                if (col == 0) {
                    float* obuf = (float*)hT[gg];
                    int row = t * 16 + hsel * 8;
                    *(float4*)&obuf[row]     = make_float4(mv[0], mv[1], mv[2], mv[3]);
                    *(float4*)&obuf[row + 4] = make_float4(mv[4], mv[5], mv[6], mv[7]);
                }
            }
        }

        // ---- coalesced 384-float stores: 3 x b128 per lane per group ----
#pragma unroll
        for (int gg = 0; gg < GRP; ++gg) {
            const float4* s4 = (const float4*)(float*)hT[gg];
            float4* d4 = (float4*)(out + (g0 + gg) * CENC);
#pragma unroll
            for (int i = 0; i < 3; ++i)
                d4[lane + i * 32] = s4[lane + i * 32];
        }
    }
}

// =====================================================================
extern "C" void kernel_launch(void* const* d_in, const int* in_sizes, int n_in,
                              void* d_out, int out_size, void* d_ws, size_t ws_size,
                              hipStream_t stream) {
    (void)in_sizes; (void)n_in; (void)out_size; (void)ws_size;
    const float* pcd = (const float*)d_in[0];
    const float* w1  = (const float*)d_in[1];
    const float* b1  = (const float*)d_in[2];
    const float* g1  = (const float*)d_in[3];
    const float* be1 = (const float*)d_in[4];
    const float* m1  = (const float*)d_in[5];
    const float* v1  = (const float*)d_in[6];
    const float* w2  = (const float*)d_in[7];
    const float* b2  = (const float*)d_in[8];
    const float* w3  = (const float*)d_in[9];
    const float* b3  = (const float*)d_in[10];
    const float* g3  = (const float*)d_in[11];
    const float* be3 = (const float*)d_in[12];
    const float* m3  = (const float*)d_in[13];
    const float* v3  = (const float*)d_in[14];
    const float* w4  = (const float*)d_in[15];
    const float* b4  = (const float*)d_in[16];

    char*  ws  = (char*)d_ws;
    float* out = (float*)d_out;

    // opt in to >64KB dynamic LDS for the encoder (WGP has 320KB)
    (void)hipFuncSetAttribute((const void*)encoder_kernel,
                              hipFuncAttributeMaxDynamicSharedMemorySize, LDS_BYTES);

    fold_convert<<<(384 * 128 + 255) / 256, 256, 0, stream>>>(
        w1, b1, g1, be1, m1, v1, w2, b2, w3, b3, g3, be3, m3, v3, w4, b4, ws);

    knn_gather<<<B_ * 8, 256, 0, stream>>>(pcd, (float*)(ws + WS_NEIGH));

    encoder_kernel<<<2048, WAVES * 32, LDS_BYTES, stream>>>((const char*)ws, out);
}
